// DAGNN_16630113370269
// MI455X (gfx1250) — matrix operand order repaired and checked by reading, not compiled, not run on hardware
//
#include <hip/hip_runtime.h>

typedef __attribute__((ext_vector_type(16))) _Float16 v16h;
typedef __attribute__((ext_vector_type(8)))  float    v8f;

#define IN_F 512
#define HID  64
#define LDS_STRIDE 68   // 64 + 4 pad to dodge bank conflicts

__device__ inline v8f wmma16(v16h a, v16h b, v8f c) {
    return __builtin_amdgcn_wmma_f32_16x16x32_f16(
        /*neg_a=*/false, a, /*neg_b=*/false, b,
        /*c_mod=*/(short)0, c, /*reuse_a=*/false, /*reuse_b=*/false);
}

// ---------------- utility kernels ----------------
__global__ void zero_kernel(float* __restrict__ p, long long n) {
    long long i = (long long)blockIdx.x * blockDim.x + threadIdx.x;
    if (i < n) p[i] = 0.f;
}

// Pack W1 (512x64) and W2 (64x64) f32 -> f16 in B-fragment lane order:
// dst[((chunk*4 + tile)*32 + lane)*16 + j] = W[(chunk*32 + (j<8 ? (lane>>4)*8 + j : 16 + (lane>>4)*8 + j-8))*64 + tile*16 + (lane&15)]
__global__ void pack_kernel(const float* __restrict__ W1, const float* __restrict__ W2,
                            _Float16* __restrict__ w1p, _Float16* __restrict__ w2p) {
    int idx = blockIdx.x * blockDim.x + threadIdx.x;
    const int n1 = 16 * 4 * 32 * 16;          // 16 K-chunks
    const int n2 = 2 * 4 * 32 * 16;           // 2 K-chunks
    if (idx < n1) {
        int j = idx & 15, l = (idx >> 4) & 31, t = (idx >> 9) & 3, kk = idx >> 11;
        int half = l >> 4, n = l & 15;
        w1p[idx] = (_Float16)W1[(kk * 32 + (j < 8 ? half * 8 + j : 16 + half * 8 + (j - 8))) * 64 + t * 16 + n];
    } else if (idx < n1 + n2) {
        int i2 = idx - n1;
        int j = i2 & 15, l = (i2 >> 4) & 31, t = (i2 >> 9) & 3, kc = i2 >> 11;
        int half = l >> 4, n = l & 15;
        w2p[i2] = (_Float16)W2[(kc * 32 + (j < 8 ? half * 8 + j : 16 + half * 8 + (j - 8))) * 64 + t * 16 + n];
    }
}

// ---------------- fused 2-layer MLP with WMMA ----------------
// 256 threads = 8 waves; each wave computes a 16-row x 64-col block of
// h0 = relu(relu(x@W1)@W2). Layer-1 output goes through wave-private LDS to
// rebuild the 16-bit A-matrix fragment layout for layer 2.
__global__ __launch_bounds__(256) void mlp_wmma_kernel(
    const float* __restrict__ x, const _Float16* __restrict__ w1p,
    const _Float16* __restrict__ w2p, float* __restrict__ h0, int N) {
    __shared__ float lds[8 * 16 * LDS_STRIDE];

    const int lane = threadIdx.x & 31;
    const int wave = threadIdx.x >> 5;
    const int half = lane >> 4;
    const int nn   = lane & 15;
    const long long rowBase = (long long)blockIdx.x * 128 + wave * 16;

    // ---- layer 1: K = 512 in 16 chunks of 32 ----
    v8f acc[4] = {v8f{}, v8f{}, v8f{}, v8f{}};
    long long arow = rowBase + nn;
    if (arow >= N) arow = N - 1;               // clamp reads; stores predicated
    const float* xr = x + arow * IN_F;
    const v16h* w1v = (const v16h*)w1p;

    for (int kk = 0; kk < 16; ++kk) {
        const int kb = kk * 32 + half * 8;
        float4 f0 = *(const float4*)(xr + kb);
        float4 f1 = *(const float4*)(xr + kb + 4);
        float4 f2 = *(const float4*)(xr + kb + 16);
        float4 f3 = *(const float4*)(xr + kb + 20);
        v16h a;
        a[0]=(_Float16)f0.x;  a[1]=(_Float16)f0.y;  a[2]=(_Float16)f0.z;  a[3]=(_Float16)f0.w;
        a[4]=(_Float16)f1.x;  a[5]=(_Float16)f1.y;  a[6]=(_Float16)f1.z;  a[7]=(_Float16)f1.w;
        a[8]=(_Float16)f2.x;  a[9]=(_Float16)f2.y;  a[10]=(_Float16)f2.z; a[11]=(_Float16)f2.w;
        a[12]=(_Float16)f3.x; a[13]=(_Float16)f3.y; a[14]=(_Float16)f3.z; a[15]=(_Float16)f3.w;
#pragma unroll
        for (int t = 0; t < 4; ++t)
            acc[t] = wmma16(a, w1v[(kk * 4 + t) * 32 + lane], acc[t]);
    }

    // relu + stash per-wave 16x64 tile in LDS (D layout -> row-major)
    float* my = lds + wave * (16 * LDS_STRIDE);
#pragma unroll
    for (int t = 0; t < 4; ++t)
#pragma unroll
        for (int r = 0; r < 8; ++r)
            my[(r + 8 * half) * LDS_STRIDE + t * 16 + nn] = fmaxf(acc[t][r], 0.f);

    // ---- layer 2: K = 64 in 2 chunks of 32 ----
    v8f d[4] = {v8f{}, v8f{}, v8f{}, v8f{}};
    const float* myr = my + nn * LDS_STRIDE;
    const v16h* w2v = (const v16h*)w2p;
#pragma unroll
    for (int kc = 0; kc < 2; ++kc) {
        const int kb = kc * 32 + half * 8;
        float4 g0 = *(const float4*)(myr + kb);
        float4 g1 = *(const float4*)(myr + kb + 4);
        float4 g2 = *(const float4*)(myr + kb + 16);
        float4 g3 = *(const float4*)(myr + kb + 20);
        v16h a;
        a[0]=(_Float16)g0.x;  a[1]=(_Float16)g0.y;  a[2]=(_Float16)g0.z;  a[3]=(_Float16)g0.w;
        a[4]=(_Float16)g1.x;  a[5]=(_Float16)g1.y;  a[6]=(_Float16)g1.z;  a[7]=(_Float16)g1.w;
        a[8]=(_Float16)g2.x;  a[9]=(_Float16)g2.y;  a[10]=(_Float16)g2.z; a[11]=(_Float16)g2.w;
        a[12]=(_Float16)g3.x; a[13]=(_Float16)g3.y; a[14]=(_Float16)g3.z; a[15]=(_Float16)g3.w;
#pragma unroll
        for (int t = 0; t < 4; ++t)
            d[t] = wmma16(a, w2v[(kc * 4 + t) * 32 + lane], d[t]);
    }

    // relu + store (predicated on node < N)
#pragma unroll
    for (int t = 0; t < 4; ++t)
#pragma unroll
        for (int r = 0; r < 8; ++r) {
            long long node = rowBase + r + 8 * half;
            if (node < N) h0[node * 64 + t * 16 + nn] = fmaxf(d[t][r], 0.f);
        }
}

// ---------------- graph kernels ----------------
__global__ void degree_kernel(const long long* __restrict__ src, const long long* __restrict__ dst,
                              float* __restrict__ degO, float* __restrict__ degI, int E) {
    int i = blockIdx.x * blockDim.x + threadIdx.x;
    if (i < E) {
        atomicAdd(&degO[(int)src[i]], 1.0f);
        atomicAdd(&degI[(int)dst[i]], 1.0f);
    }
}

__global__ void dinv_kernel(float* __restrict__ deg, int n) {
    int i = blockIdx.x * blockDim.x + threadIdx.x;
    if (i < n) {
        float d = deg[i];
        deg[i] = (d > 0.f) ? rsqrtf(d) : 0.f;   // counts >= 1 when > 0
    }
}

__global__ void norm_kernel(const long long* __restrict__ src, const long long* __restrict__ dst,
                            const float* __restrict__ dO, const float* __restrict__ dI,
                            float* __restrict__ nrm, int E) {
    int i = blockIdx.x * blockDim.x + threadIdx.x;
    if (i < E) nrm[i] = dO[(int)src[i]] * dI[(int)dst[i]];
}

// One (edge, feature-pair) per thread: gather float2, scale, atomic scatter-add.
__global__ void scatter_kernel(const long long* __restrict__ src, const long long* __restrict__ dst,
                               const float* __restrict__ nrm, const float* __restrict__ fin,
                               float* __restrict__ fout, int E) {
    long long gid = (long long)blockIdx.x * blockDim.x + threadIdx.x;
    int e = (int)(gid >> 5);
    if (e >= E) return;
    int j = ((int)gid & 31) * 2;
    int s = (int)src[e], d2 = (int)dst[e];
    float w = nrm[e];
    float2 v = *(const float2*)(fin + (long long)s * 64 + j);
    atomicAdd(fout + (long long)d2 * 64 + j,     v.x * w);
    atomicAdd(fout + (long long)d2 * 64 + j + 1, v.y * w);
}

// out[n,:] += sigmoid(feat[n,:] . s) * feat[n,:]   (one wave per node)
__global__ __launch_bounds__(256) void gate_kernel(const float* __restrict__ feat,
                                                   const float* __restrict__ s,
                                                   float* __restrict__ out, int N) {
    int lane = threadIdx.x & 31;
    int wave = threadIdx.x >> 5;
    long long n = (long long)blockIdx.x * 8 + wave;
    if (n >= N) return;
    const float* f = feat + n * 64;
    float f0 = f[lane], f1 = f[lane + 32];
    float p = f0 * s[lane] + f1 * s[lane + 32];
#pragma unroll
    for (int m = 16; m > 0; m >>= 1) p += __shfl_xor(p, m, 32);
    float g = 1.f / (1.f + __expf(-p));
    out[n * 64 + lane]      += g * f0;
    out[n * 64 + lane + 32] += g * f1;
}

// ---------------- host launcher ----------------
extern "C" void kernel_launch(void* const* d_in, const int* in_sizes, int n_in,
                              void* d_out, int out_size, void* d_ws, size_t ws_size,
                              hipStream_t stream) {
    const float*     x   = (const float*)d_in[0];
    const long long* ei  = (const long long*)d_in[1];
    const float*     W1  = (const float*)d_in[2];
    const float*     W2  = (const float*)d_in[3];
    const float*     sv  = (const float*)d_in[4];
    float*           out = (float*)d_out;

    const int N = in_sizes[0] / IN_F;
    const int E = in_sizes[1] / 2;
    const long long* srcI = ei;
    const long long* dstI = ei + E;

    char* ws = (char*)d_ws;
    size_t off = 0;
    auto take = [&](size_t bytes) -> char* {
        char* p = ws + off;
        off = (off + bytes + 255) & ~(size_t)255;
        return p;
    };
    _Float16* w1p  = (_Float16*)take((size_t)16 * 4 * 32 * 16 * 2);
    _Float16* w2p  = (_Float16*)take((size_t)2 * 4 * 32 * 16 * 2);
    float* degO    = (float*)take((size_t)N * 4);
    float* degI    = (float*)take((size_t)N * 4);
    float* nrm     = (float*)take((size_t)E * 4);
    float* featA   = (float*)take((size_t)N * 64 * 4);
    float* featB   = (float*)take((size_t)N * 64 * 4);

    const long long nFeat = (long long)N * 64;
    const int zbF = (int)((nFeat + 255) / 256);
    const int zbN = (N + 255) / 256;

    // fresh state every launch (graph-replay safe)
    zero_kernel<<<zbF, 256, 0, stream>>>(out, nFeat);
    zero_kernel<<<zbN, 256, 0, stream>>>(degO, N);
    zero_kernel<<<zbN, 256, 0, stream>>>(degI, N);

    pack_kernel<<<(16 * 4 * 32 * 16 + 2 * 4 * 32 * 16 + 255) / 256, 256, 0, stream>>>(W1, W2, w1p, w2p);
    mlp_wmma_kernel<<<(N + 127) / 128, 256, 0, stream>>>(x, w1p, w2p, featA, N);

    degree_kernel<<<(E + 255) / 256, 256, 0, stream>>>(srcI, dstI, degO, degI, E);
    dinv_kernel<<<zbN, 256, 0, stream>>>(degO, N);
    dinv_kernel<<<zbN, 256, 0, stream>>>(degI, N);
    norm_kernel<<<(E + 255) / 256, 256, 0, stream>>>(srcI, dstI, degO, degI, nrm, E);

    // hop 0 gate
    gate_kernel<<<(N + 7) / 8, 256, 0, stream>>>(featA, sv, out, N);

    // K = 10 propagation hops, gate fused after each
    float* cur = featA;
    float* nxt = featB;
    const long long scatThreads = (long long)E * 32;
    const int scatBlocks = (int)((scatThreads + 255) / 256);
    for (int k = 0; k < 10; ++k) {
        zero_kernel<<<zbF, 256, 0, stream>>>(nxt, nFeat);
        scatter_kernel<<<scatBlocks, 256, 0, stream>>>(srcI, dstI, nrm, cur, nxt, E);
        gate_kernel<<<(N + 7) / 8, 256, 0, stream>>>(nxt, sv, out, N);
        float* t = cur; cur = nxt; nxt = t;
    }
}
